// FactorAtt_ConvRelPosEnc_80564996538437
// MI455X (gfx1250) — compile-verified, matrix operand order checked by reading, not run on hardware
//
#include <hip/hip_runtime.h>

// B=8, N=4096, C=768, H=12, D=64
#define NB 8
#define NSEQ 4096
#define NC 768
#define NH 12
#define ND 64
#define PHEAD (25165824L)   // B*H*N*D = B*N*C elements

typedef __attribute__((ext_vector_type(16))) __bf16 v16bf;
typedef __attribute__((ext_vector_type(8)))  __bf16 bf16x8;
typedef __attribute__((ext_vector_type(8)))  float  v8f;
typedef __attribute__((ext_vector_type(4)))  float  f32x4;

// ---- WMMA operand fragment loaders (ISA 7.12.2, 16-bit A/B 16x32) ----------
// lane L -> row L%16, K base (L/16)*8, elems [0..7]=K+0..7, [8..15]=K+16..23.

// From row-major f32 (weights): load f32, convert inline.
__device__ __forceinline__ v16bf load_frag_f32(const float* __restrict__ p0,
                                               int ld, int lane) {
  const float* p = p0 + (long)(lane & 15) * ld + ((lane >> 4) << 3);
  f32x4 a0 = *(const f32x4*)(p + 0);
  f32x4 a1 = *(const f32x4*)(p + 4);
  f32x4 c0 = *(const f32x4*)(p + 16);
  f32x4 c1 = *(const f32x4*)(p + 20);
  v16bf r;
  r[0]  = (__bf16)a0[0]; r[1]  = (__bf16)a0[1]; r[2]  = (__bf16)a0[2]; r[3]  = (__bf16)a0[3];
  r[4]  = (__bf16)a1[0]; r[5]  = (__bf16)a1[1]; r[6]  = (__bf16)a1[2]; r[7]  = (__bf16)a1[3];
  r[8]  = (__bf16)c0[0]; r[9]  = (__bf16)c0[1]; r[10] = (__bf16)c0[2]; r[11] = (__bf16)c0[3];
  r[12] = (__bf16)c1[0]; r[13] = (__bf16)c1[1]; r[14] = (__bf16)c1[2]; r[15] = (__bf16)c1[3];
  return r;
}

// From row-major bf16 (activations): direct b128 loads, no conversion.
__device__ __forceinline__ v16bf load_frag_bf16(const __bf16* __restrict__ p0,
                                                int ld, int lane) {
  const __bf16* p = p0 + (long)(lane & 15) * ld + ((lane >> 4) << 3);
  bf16x8 lo = *(const bf16x8*)(p);        // K+0..7   (16B aligned)
  bf16x8 hi = *(const bf16x8*)(p + 16);   // K+16..23
  v16bf r;
#pragma unroll
  for (int i = 0; i < 8; ++i) { r[i] = lo[i]; r[8 + i] = hi[i]; }
  return r;
}

__device__ __forceinline__ v8f wmma_bf16(v16bf a, v16bf b, v8f c) {
  return __builtin_amdgcn_wmma_f32_16x16x32_bf16(
      false, a, false, b, (short)0, c, false, false);
}

// D tile (16x16 f32): VGPR r -> row r + (lane/16)*8, col lane%16.
__device__ __forceinline__ void store_tile_f32(float* __restrict__ base, int ldo,
                                               v8f d, int lane) {
  const int row = (lane >> 4) << 3;
  const int col = lane & 15;
#pragma unroll
  for (int r = 0; r < 8; ++r) base[(long)(row + r) * ldo + col] = d[r];
}

__device__ __forceinline__ void store_tile_bf16(__bf16* __restrict__ base, int ldo,
                                                v8f d, int lane) {
  const int row = (lane >> 4) << 3;
  const int col = lane & 15;
#pragma unroll
  for (int r = 0; r < 8; ++r)
    base[(long)(row + r) * ldo + col] = (__bf16)d[r];
}

// ---------------- Kernel 1: input projection -> bf16, head-split ------------
// Out(B,H,N,D)[bh][n][d] = sum_i X[b*N+n][i] * W[h*64+d][i]
__global__ void proj_kernel(const float* __restrict__ X,
                            const float* __restrict__ W,
                            __bf16* __restrict__ Out) {
  const int lane = threadIdx.x & 31;
  const int tileId = blockIdx.x * 8 + (threadIdx.x >> 5);
  const int tn = tileId % (NC / 32);
  const long r0 = (long)(tileId / (NC / 32)) * 32;
  const int c0 = tn * 32;

  v8f acc00 = {}, acc01 = {}, acc10 = {}, acc11 = {};
#pragma unroll 4
  for (int kk = 0; kk < NC; kk += 32) {
    v16bf a0 = load_frag_f32(X + r0 * NC + kk, NC, lane);
    v16bf a1 = load_frag_f32(X + (r0 + 16) * NC + kk, NC, lane);
    v16bf b0 = load_frag_f32(W + (long)c0 * NC + kk, NC, lane);
    v16bf b1 = load_frag_f32(W + (long)(c0 + 16) * NC + kk, NC, lane);
    acc00 = wmma_bf16(a0, b0, acc00);
    acc01 = wmma_bf16(a0, b1, acc01);
    acc10 = wmma_bf16(a1, b0, acc10);
    acc11 = wmma_bf16(a1, b1, acc11);
  }
  const int b = (int)(r0 >> 12), n0 = (int)(r0 & (NSEQ - 1));
  const int h = c0 >> 6, d0 = c0 & 63;
  __bf16* ob = Out + (((long)b * NH + h) * NSEQ + n0) * ND + d0;
  store_tile_bf16(ob, ND, acc00, lane);
  store_tile_bf16(ob + 16, ND, acc01, lane);
  store_tile_bf16(ob + 16L * ND, ND, acc10, lane);
  store_tile_bf16(ob + 16L * ND + 16, ND, acc11, lane);
}

// -------- Kernel 2: fused column-softmax + ctx^T, one block per (b,h) -------
// ctxT[bh][v][k] = (1/Z_k) * sum_n exp(kh[bh][n][k]-m_k) * vh[bh][n][v]
__global__ void ctx_softmax_kernel(const __bf16* __restrict__ kh,
                                   const __bf16* __restrict__ vh,
                                   __bf16* __restrict__ ctxT) {
  __shared__ float red[256];
  __shared__ float colmax[64];
  __shared__ float colinv[64];
  __shared__ float Ks[64][65];
  __shared__ float Vs[64][65];

  const long base = (long)blockIdx.x * (long)NSEQ * ND;
  const int d = threadIdx.x & 63;
  const int ty = threadIdx.x >> 6;  // 0..3

  // column max over n
  float mx = -3.0e38f;
  for (int n = ty; n < NSEQ; n += 4)
    mx = fmaxf(mx, (float)kh[base + (long)n * ND + d]);
  red[threadIdx.x] = mx;
  __syncthreads();
  if (ty == 0)
    colmax[d] = fmaxf(fmaxf(red[d], red[d + 64]), fmaxf(red[d + 128], red[d + 192]));
  __syncthreads();
  mx = colmax[d];
  __syncthreads();

  // column sum of exp
  float s = 0.0f;
  for (int n = ty; n < NSEQ; n += 4)
    s += __expf((float)kh[base + (long)n * ND + d] - mx);
  red[threadIdx.x] = s;
  __syncthreads();
  if (ty == 0)
    colinv[d] = 1.0f / (red[d] + red[d + 64] + red[d + 128] + red[d + 192]);
  __syncthreads();

  // accumulate exp-weighted outer products through LDS tiles
  const int tx = threadIdx.x & 15;   // v group (4 cols)
  const int tk = threadIdx.x >> 4;   // k group (4 rows)
  float acc[4][4] = {};
  for (int nt = 0; nt < NSEQ; nt += 64) {
    for (int i = threadIdx.x; i < 64 * 64; i += 256) {
      const int n = i >> 6, c = i & 63;
      const long idx = base + (long)(nt + n) * ND + c;
      Ks[n][c] = __expf((float)kh[idx] - colmax[c]);
      Vs[n][c] = (float)vh[idx];
    }
    __syncthreads();
#pragma unroll 4
    for (int n = 0; n < 64; ++n) {
      float kr[4], vr[4];
#pragma unroll
      for (int i = 0; i < 4; ++i) kr[i] = Ks[n][tk * 4 + i];
#pragma unroll
      for (int j = 0; j < 4; ++j) vr[j] = Vs[n][tx * 4 + j];
#pragma unroll
      for (int i = 0; i < 4; ++i)
#pragma unroll
        for (int j = 0; j < 4; ++j) acc[i][j] = fmaf(kr[i], vr[j], acc[i][j]);
    }
    __syncthreads();
  }
  __bf16* ob = ctxT + (long)blockIdx.x * ND * ND;
#pragma unroll
  for (int i = 0; i < 4; ++i) {
    const float inv = colinv[tk * 4 + i];
#pragma unroll
    for (int j = 0; j < 4; ++j)
      ob[(long)(tx * 4 + j) * ND + (tk * 4 + i)] = (__bf16)(acc[i][j] * inv);
  }
}

// ---------------- Kernel 3: factor_att = qh @ ctx, batched over (b,h) -------
// xfa[(b*N+n)*C + h*64 + v] = sum_k qh[bh][n][k] * ctxT[bh][v][k]
__global__ void fa_kernel(const __bf16* __restrict__ qh,
                          const __bf16* __restrict__ ctxT,
                          __bf16* __restrict__ xfa) {
  const int lane = threadIdx.x & 31;
  const int bh = blockIdx.y;
  const int tileId = blockIdx.x * 8 + (threadIdx.x >> 5);  // 0..255
  const int tn = tileId & 1;
  const long r0 = (long)(tileId >> 1) * 32;
  const int c0 = tn * 32;

  const __bf16* A = qh + (long)bh * NSEQ * ND;
  const __bf16* Bm = ctxT + (long)bh * ND * ND;

  v8f acc00 = {}, acc01 = {}, acc10 = {}, acc11 = {};
#pragma unroll
  for (int kk = 0; kk < ND; kk += 32) {
    v16bf a0 = load_frag_bf16(A + r0 * ND + kk, ND, lane);
    v16bf a1 = load_frag_bf16(A + (r0 + 16) * ND + kk, ND, lane);
    v16bf b0 = load_frag_bf16(Bm + (long)c0 * ND + kk, ND, lane);
    v16bf b1 = load_frag_bf16(Bm + (long)(c0 + 16) * ND + kk, ND, lane);
    acc00 = wmma_bf16(a0, b0, acc00);
    acc01 = wmma_bf16(a0, b1, acc01);
    acc10 = wmma_bf16(a1, b0, acc10);
    acc11 = wmma_bf16(a1, b1, acc11);
  }
  const int b = bh / NH, h = bh % NH;
  __bf16* ob = xfa + ((long)b * NSEQ + r0) * NC + h * ND + c0;
  store_tile_bf16(ob, NC, acc00, lane);
  store_tile_bf16(ob + 16, NC, acc01, lane);
  store_tile_bf16(ob + 16L * NC, NC, acc10, lane);
  store_tile_bf16(ob + 16L * NC + 16, NC, acc11, lane);
}

// ---------------- Kernel 4: output projection (bf16 A) + bias -> f32 --------
__global__ void outproj_kernel(const __bf16* __restrict__ X,
                               const float* __restrict__ W,
                               const float* __restrict__ bias,
                               float* __restrict__ Out) {
  const int lane = threadIdx.x & 31;
  const int tileId = blockIdx.x * 8 + (threadIdx.x >> 5);
  const int tn = tileId % (NC / 32);
  const long r0 = (long)(tileId / (NC / 32)) * 32;
  const int c0 = tn * 32;

  v8f acc00 = {}, acc01 = {}, acc10 = {}, acc11 = {};
#pragma unroll 4
  for (int kk = 0; kk < NC; kk += 32) {
    v16bf a0 = load_frag_bf16(X + r0 * NC + kk, NC, lane);
    v16bf a1 = load_frag_bf16(X + (r0 + 16) * NC + kk, NC, lane);
    v16bf b0 = load_frag_f32(W + (long)c0 * NC + kk, NC, lane);
    v16bf b1 = load_frag_f32(W + (long)(c0 + 16) * NC + kk, NC, lane);
    acc00 = wmma_bf16(a0, b0, acc00);
    acc01 = wmma_bf16(a0, b1, acc01);
    acc10 = wmma_bf16(a1, b0, acc10);
    acc11 = wmma_bf16(a1, b1, acc11);
  }
  const int col = lane & 15;
  const float bz0 = bias[c0 + col];
  const float bz1 = bias[c0 + 16 + col];
#pragma unroll
  for (int r = 0; r < 8; ++r) {
    acc00[r] += bz0; acc10[r] += bz0;
    acc01[r] += bz1; acc11[r] += bz1;
  }
  float* ob = Out + r0 * NC + c0;
  store_tile_f32(ob, NC, acc00, lane);
  store_tile_f32(ob + 16, NC, acc01, lane);
  store_tile_f32(ob + 16L * NC, NC, acc10, lane);
  store_tile_f32(ob + 16L * NC + 16, NC, acc11, lane);
}

extern "C" void kernel_launch(void* const* d_in, const int* in_sizes, int n_in,
                              void* d_out, int out_size, void* d_ws, size_t ws_size,
                              hipStream_t stream) {
  const float* q  = (const float*)d_in[0];
  const float* k  = (const float*)d_in[1];
  const float* v  = (const float*)d_in[2];
  const float* Wq = (const float*)d_in[3];
  const float* Wk = (const float*)d_in[4];
  const float* Wv = (const float*)d_in[5];
  const float* Wp = (const float*)d_in[6];
  const float* bp = (const float*)d_in[7];

  __bf16* ws   = (__bf16*)d_ws;
  __bf16* qh   = ws;              // (B,H,N,D) bf16
  __bf16* kh   = ws + PHEAD;      // (B,H,N,D) bf16 (raw projection; softmax fused later)
  __bf16* vh   = ws + 2 * PHEAD;  // (B,H,N,D) bf16
  __bf16* xfa  = ws + 3 * PHEAD;  // (B,N,C)   bf16
  __bf16* ctxT = ws + 4 * PHEAD;  // (BH, D(v), D(k)) bf16

  const dim3 blk(256);
  const int projBlocks = (32768 / 32) * (NC / 32) / 8;  // 3072

  proj_kernel<<<projBlocks, blk, 0, stream>>>(q, Wq, qh);
  proj_kernel<<<projBlocks, blk, 0, stream>>>(k, Wk, kh);
  proj_kernel<<<projBlocks, blk, 0, stream>>>(v, Wv, vh);
  ctx_softmax_kernel<<<NB * NH, blk, 0, stream>>>(kh, vh, ctxT);
  fa_kernel<<<dim3((NSEQ / 32) * (ND / 32) / 8, NB * NH), blk, 0, stream>>>(qh, ctxT, xfa);
  outproj_kernel<<<projBlocks, blk, 0, stream>>>(xfa, Wp, bp, (float*)d_out);
}
